// multi_loss_69200513073925
// MI455X (gfx1250) — matrix operand order repaired and checked by reading, not compiled, not run on hardware
//
#include <hip/hip_runtime.h>
#include <hip/hip_bf16.h>
#include <math.h>

typedef _Float16 v16h __attribute__((ext_vector_type(16)));
typedef float    v8f  __attribute__((ext_vector_type(8)));

// ws layout (doubles): [0]=Sum t^2  [1]=Sum p^2  [2]=Sum t*p  [3]=Sum (t-p)^2  [4]=rank sum
__global__ void init_ws_kernel(double* ws) {
    int i = threadIdx.x;
    if (i < 8) ws[i] = 0.0;
}

// One full wave computes the Gram reductions with f16-split (Dekker) WMMA.
// A rows 0..3 = {t_hi, t_lo, p_hi, p_lo} over K=32 chunk; B cols 0..3 = same.
// D[r][c] accumulates cross dot products in f32; hi/lo recombination gives
// f32-grade Sum t^2, Sum p^2, Sum t*p.
__global__ void __launch_bounds__(32) gram_wmma_kernel(const float* __restrict__ t,
                                                       const float* __restrict__ p,
                                                       double* ws, int B) {
    __shared__ _Float16 sh[4][32];   // 0:t_hi 1:t_lo 2:p_hi 3:p_lo
    __shared__ float    dred[4][4];  // D[m][n], m,n in 0..3
    const int L    = threadIdx.x;          // lane 0..31
    const int r    = L & 15;               // A row / B col index
    const int ka   = (L >= 16) ? 8  : 0;   // ISA 7.12.2 table K blocks; unused (A uses koff)
    const int kb   = (L >= 16) ? 24 : 16;
    const int koff = (L >= 16) ? 16 : 0;   // B layout: K offset per half-wave

    v8f c = {};
    for (int k0 = 0; k0 < B; k0 += 32) {
        const int idx = k0 + L;
        float tv = (idx < B) ? t[idx] : 0.0f;
        float pv = (idx < B) ? p[idx] : 0.0f;
        _Float16 th = (_Float16)tv;
        _Float16 tl = (_Float16)(tv - (float)th);
        _Float16 ph = (_Float16)pv;
        _Float16 pl = (_Float16)(pv - (float)ph);
        __syncthreads();
        sh[0][L] = th; sh[1][L] = tl; sh[2][L] = ph; sh[3][L] = pl;
        __syncthreads();

        v16h a = {};
        v16h b = {};
        if (r < 4) {
#pragma unroll
            for (int h = 0; h < 8; ++h) {       // A: 16-bit 16x32 layout
                a[h]     = sh[r][koff + h];
                a[8 + h] = sh[r][koff + 8 + h];
            }
#pragma unroll
            for (int h = 0; h < 16; ++h) {      // B: 32x16 layout
                b[h] = sh[r][koff + h];
            }
        }
        // (neg_a, A, neg_b, B, c_mod, C, reuse_a, reuse_b)
        c = __builtin_amdgcn_wmma_f32_16x16x32_f16(false, a, false, b,
                                                   (short)0, c, false, false);
    }

    // C/D layout: lane n (0..15) col N=n, VGPR m holds M=m  -> need lanes 0..3, c[0..3]
    if (L < 4) {
#pragma unroll
        for (int m = 0; m < 4; ++m) dred[m][L] = c[m];
    }
    __syncthreads();
    if (L == 0) {
        double tt = (double)dred[0][0] + 2.0 * (double)dred[0][1] + (double)dred[1][1];
        double pp = (double)dred[2][2] + 2.0 * (double)dred[2][3] + (double)dred[3][3];
        double tp = (double)dred[0][2] + (double)dred[0][3]
                  + (double)dred[1][2] + (double)dred[1][3];
        double sq = tt - 2.0 * tp + pp;   // Sum (t-p)^2; expanded form, can cancel when t ~ p
        ws[0] = tt; ws[1] = pp; ws[2] = tp; ws[3] = sq;
    }
}

// Pairwise margin-ranking loss over strict upper triangle, 256x256 tiles.
// loss(i<j): t_i>t_j -> relu(p_j-p_i); t_i<t_j -> relu(p_i-p_j); tie -> |p_i-p_j|
__global__ void __launch_bounds__(256) rank_kernel(const float* __restrict__ t,
                                                   const float* __restrict__ p,
                                                   double* ws_rank, int B) {
    if (blockIdx.y < blockIdx.x) return;   // only tiles intersecting i<j
    __shared__ float  sjt[256];
    __shared__ float  sjp[256];
    __shared__ double red[256];

    const int tid = threadIdx.x;
    const int i0  = blockIdx.x * 256;
    const int j0  = blockIdx.y * 256;

    const int j = j0 + tid;
    sjt[tid] = (j < B) ? t[j] : 0.0f;
    sjp[tid] = (j < B) ? p[j] : 0.0f;
    __syncthreads();

    const int i = i0 + tid;
    float acc = 0.0f;
    if (i < B) {
        const float ti = t[i];
        const float pi = p[i];
#pragma unroll 8
        for (int jj = 0; jj < 256; ++jj) {
            const int gj = j0 + jj;
            if (gj < B && gj > i) {
                const float dt = ti - sjt[jj];
                const float dp = pi - sjp[jj];
                const float l  = (dt > 0.0f) ? fmaxf(0.0f, -dp)
                               : (dt < 0.0f) ? fmaxf(0.0f,  dp)
                                             : fabsf(dp);
                acc += l;
            }
        }
    }
    red[tid] = (double)acc;
    __syncthreads();
    for (int s = 128; s > 0; s >>= 1) {
        if (tid < s) red[tid] += red[tid + s];
        __syncthreads();
    }
    if (tid == 0) atomicAdd(ws_rank, red[0]);  // global_atomic_add_f64
}

__global__ void finalize_kernel(const double* ws, float* out, int B) {
    const double tt = ws[0], pp = ws[1], tp = ws[2], sq = ws[3], rank = ws[4];
    const double mse = sq / (double)B;
    double denom = sqrt(tt) * sqrt(pp);
    if (denom < 1e-8) denom = 1e-8;
    double sim = 1.0 - tp / denom;
    if (sim < 0.0) sim = 0.0;
    out[0] = (float)(40.0 * mse + 100.0 * sim + 1.0 * rank);
}

extern "C" void kernel_launch(void* const* d_in, const int* in_sizes, int n_in,
                              void* d_out, int out_size, void* d_ws, size_t ws_size,
                              hipStream_t stream) {
    const float* t = (const float*)d_in[0];   // y_trues, float32 [B]
    const float* p = (const float*)d_in[1];   // y_preds, float32 [B]
    float*  out = (float*)d_out;              // scalar float32
    double* ws  = (double*)d_ws;              // >= 8 doubles
    const int B = in_sizes[0];

    init_ws_kernel<<<1, 8, 0, stream>>>(ws);
    gram_wmma_kernel<<<1, 32, 0, stream>>>(t, p, ws, B);

    const int nt = (B + 255) / 256;
    dim3 grid(nt, nt);
    rank_kernel<<<grid, 256, 0, stream>>>(t, p, ws + 4, B);

    finalize_kernel<<<1, 1, 0, stream>>>(ws, out, B);
}
